// CGCNN_30605936951683
// MI455X (gfx1250) — compile-verified, matrix operand order checked by reading, not run on hardware
//
#include <hip/hip_runtime.h>

typedef __attribute__((ext_vector_type(16))) __bf16 v16bf;
typedef __attribute__((ext_vector_type(8)))  float  v8f;

#define N_NODES 100000
#define N_EDGES 3200000
#define F_IN    9
#define F_E     16
#define NGRAPH  256
#define ZI      34          // 2*F_IN + F_E
#define ZPAD    40          // padded z row in LDS (cols 34..39 zero)
#define OUTD    32

__device__ __forceinline__ __bf16 bf_hi(float v) {
    return (__bf16)v;
}
__device__ __forceinline__ __bf16 bf_lo(float v) {
    __bf16 h = (__bf16)v;
    return (__bf16)(v - (float)h);
}

// ---------------------------------------------------------------------------
// Edge kernel: 16 edges per wave per tile.
//   msg = sigmoid(z@Wf + bf) * softplus(z@Ws + bs), atomically added to agg[dst].
// Dense math on v_wmma_f32_16x16x32_bf16 with bf16x3 split (fp32-grade).
// ---------------------------------------------------------------------------
__global__ void __launch_bounds__(256)
cgconv_edge_wmma(const float* __restrict__ hin, const int* __restrict__ eidx,
                 const float* __restrict__ ea,
                 const float* __restrict__ Wf, const float* __restrict__ Ws,
                 const float* __restrict__ bfp, const float* __restrict__ bsp,
                 float* __restrict__ agg)
{
    __shared__ float zf[8][16 * ZPAD];   // per-wave z tiles (fp32)
    __shared__ int   sdst[8][16];        // per-wave dst indices

    const int  lane = threadIdx.x & 31;
    const int  wv   = threadIdx.x >> 5;
    const int  n    = lane & 15;         // output column (N dim), valid if < 9
    const bool hiL  = lane >= 16;

    // ---- B fragments (weights), built once per wave. bf16 hi/lo split. ----
    // Layout (16-bit 16x32 A-analogue): lanes 0-15 hold K{0..7,16..23},
    // lanes 16-31 hold K{8..15,24..31}; chunk1 covers K=32..63 (only 32,33 real).
    v16bf Bf0h, Bf0l, Bf1h, Bf1l, Bs0h, Bs0l, Bs1h, Bs1l;
    #pragma unroll
    for (int j = 0; j < 16; ++j) {
        Bf0h[j] = (__bf16)0.f; Bf0l[j] = (__bf16)0.f;
        Bf1h[j] = (__bf16)0.f; Bf1l[j] = (__bf16)0.f;
        Bs0h[j] = (__bf16)0.f; Bs0l[j] = (__bf16)0.f;
        Bs1h[j] = (__bf16)0.f; Bs1l[j] = (__bf16)0.f;
    }
    if (n < F_IN) {
        #pragma unroll
        for (int j = 0; j < 8; ++j) {
            const int k0 = (hiL ? 8 : 0) + j;     // frag elems 0..7
            const int k1 = (hiL ? 24 : 16) + j;   // frag elems 8..15
            const float wf0 = Wf[k0 * F_IN + n];
            const float wf1 = Wf[k1 * F_IN + n];
            const float ws0 = Ws[k0 * F_IN + n];
            const float ws1 = Ws[k1 * F_IN + n];
            Bf0h[j]     = bf_hi(wf0);  Bf0l[j]     = bf_lo(wf0);
            Bf0h[8 + j] = bf_hi(wf1);  Bf0l[8 + j] = bf_lo(wf1);
            Bs0h[j]     = bf_hi(ws0);  Bs0l[j]     = bf_lo(ws0);
            Bs0h[8 + j] = bf_hi(ws1);  Bs0l[8 + j] = bf_lo(ws1);
        }
        if (!hiL) {  // chunk1 elems 0..7 <-> K=32..39; only K=32,33 exist
            const float wf32 = Wf[32 * F_IN + n], wf33 = Wf[33 * F_IN + n];
            const float ws32 = Ws[32 * F_IN + n], ws33 = Ws[33 * F_IN + n];
            Bf1h[0] = bf_hi(wf32); Bf1l[0] = bf_lo(wf32);
            Bf1h[1] = bf_hi(wf33); Bf1l[1] = bf_lo(wf33);
            Bs1h[0] = bf_hi(ws32); Bs1l[0] = bf_lo(ws32);
            Bs1h[1] = bf_hi(ws33); Bs1l[1] = bf_lo(ws33);
        }
    }
    const float biasf = (n < F_IN) ? bfp[n] : 0.f;
    const float biass = (n < F_IN) ? bsp[n] : 0.f;

    const int ntiles       = (N_EDGES + 15) >> 4;
    const int tilesPerIter = gridDim.x * 8;
    const int iters        = (ntiles + tilesPerIter - 1) / tilesPerIter;

    for (int it = 0; it < iters; ++it) {
        const int  tile  = (it * gridDim.x + blockIdx.x) * 8 + wv;
        const bool valid = tile < ntiles;
        const int  e0    = tile << 4;

        if (valid) {
            if (!hiL) {                                   // h[dst] -> z[0..8]
                const int m = lane;
                const int e = min(e0 + m, N_EDGES - 1);
                const int d = eidx[N_EDGES + e];
                sdst[wv][m] = d;
                const float* hp = hin + (size_t)d * F_IN;
                float* zr = &zf[wv][m * ZPAD];
                #pragma unroll
                for (int k = 0; k < F_IN; ++k) zr[k] = hp[k];
                #pragma unroll
                for (int k = ZI; k < ZPAD; ++k) zr[k] = 0.f;   // K pad
            } else {                                      // h[src] -> z[9..17]
                const int m = lane - 16;
                const int e = min(e0 + m, N_EDGES - 1);
                const int s = eidx[e];
                const float* hp = hin + (size_t)s * F_IN;
                float* zr = &zf[wv][m * ZPAD];
                #pragma unroll
                for (int k = 0; k < F_IN; ++k) zr[F_IN + k] = hp[k];
            }
            // edge_attr -> z[18..33]; 16x16 contiguous block, 64B-aligned
            {
                const float4* eap = reinterpret_cast<const float4*>(ea + (size_t)e0 * F_E);
                float4 u0 = eap[lane * 2 + 0];
                float4 u1 = eap[lane * 2 + 1];
                float* zr = &zf[wv][(lane >> 1) * ZPAD + 18 + ((lane & 1) << 3)];
                zr[0] = u0.x; zr[1] = u0.y; zr[2] = u0.z; zr[3] = u0.w;
                zr[4] = u1.x; zr[5] = u1.y; zr[6] = u1.z; zr[7] = u1.w;
            }
        }
        __syncthreads();

        if (valid) {
            // ---- A fragments from LDS (bf16 hi/lo split) ----
            const int m = lane & 15;
            const float* zr = &zf[wv][m * ZPAD];
            v16bf a0h, a0l, a1h, a1l;
            #pragma unroll
            for (int j = 0; j < 8; ++j) {
                const float z0 = zr[(hiL ? 8 : 0) + j];
                const float z1 = zr[(hiL ? 24 : 16) + j];
                a0h[j]     = bf_hi(z0);  a0l[j]     = bf_lo(z0);
                a0h[8 + j] = bf_hi(z1);  a0l[8 + j] = bf_lo(z1);
                a1h[8 + j] = (__bf16)0.f; a1l[8 + j] = (__bf16)0.f;
                if (!hiL) {
                    const float z2 = zr[32 + j];
                    a1h[j] = bf_hi(z2); a1l[j] = bf_lo(z2);
                } else {
                    a1h[j] = (__bf16)0.f; a1l[j] = (__bf16)0.f;
                }
            }

            v8f cf = {};
            cf = __builtin_amdgcn_wmma_f32_16x16x32_bf16(false, a0h, false, Bf0h, (short)0, cf, false, false);
            cf = __builtin_amdgcn_wmma_f32_16x16x32_bf16(false, a0h, false, Bf0l, (short)0, cf, false, false);
            cf = __builtin_amdgcn_wmma_f32_16x16x32_bf16(false, a0l, false, Bf0h, (short)0, cf, false, false);
            cf = __builtin_amdgcn_wmma_f32_16x16x32_bf16(false, a1h, false, Bf1h, (short)0, cf, false, false);
            cf = __builtin_amdgcn_wmma_f32_16x16x32_bf16(false, a1h, false, Bf1l, (short)0, cf, false, false);
            cf = __builtin_amdgcn_wmma_f32_16x16x32_bf16(false, a1l, false, Bf1h, (short)0, cf, false, false);

            v8f cs = {};
            cs = __builtin_amdgcn_wmma_f32_16x16x32_bf16(false, a0h, false, Bs0h, (short)0, cs, false, false);
            cs = __builtin_amdgcn_wmma_f32_16x16x32_bf16(false, a0h, false, Bs0l, (short)0, cs, false, false);
            cs = __builtin_amdgcn_wmma_f32_16x16x32_bf16(false, a0l, false, Bs0h, (short)0, cs, false, false);
            cs = __builtin_amdgcn_wmma_f32_16x16x32_bf16(false, a1h, false, Bs1h, (short)0, cs, false, false);
            cs = __builtin_amdgcn_wmma_f32_16x16x32_bf16(false, a1h, false, Bs1l, (short)0, cs, false, false);
            cs = __builtin_amdgcn_wmma_f32_16x16x32_bf16(false, a1l, false, Bs1h, (short)0, cs, false, false);

            // C/D layout: lanes 0-15 -> M = r, lanes 16-31 -> M = r + 8, N = lane%16
            if (n < F_IN) {
                const int mbase = hiL ? 8 : 0;
                #pragma unroll
                for (int r = 0; r < 8; ++r) {
                    const int mm = mbase + r;
                    const int e  = e0 + mm;
                    if (e < N_EDGES) {
                        const float gate = cf[r] + biasf;
                        const float soft = cs[r] + biass;
                        const float sg = 1.f / (1.f + expf(-gate));
                        const float sp = (soft > 20.f) ? soft : log1pf(expf(soft));
                        const int d = sdst[wv][mm];
                        atomicAdd(&agg[(size_t)d * F_IN + n], sg * sp);
                    }
                }
            }
        }
        __syncthreads();
    }
}

// ---------------------------------------------------------------------------
// Helper kernels
// ---------------------------------------------------------------------------
__global__ void init_enc_cnt_kernel(float* enc, int* cnt) {
    int i = blockIdx.x * blockDim.x + threadIdx.x;
    if (i < NGRAPH * 27) enc[i] = 0.f;
    if (i < NGRAPH)      cnt[i] = 0;
}

__global__ void count_kernel(const int* __restrict__ bat, int* cnt) {
    int i = blockIdx.x * blockDim.x + threadIdx.x;
    if (i < N_NODES) atomicAdd(&cnt[bat[i]], 1);
}

__global__ void layer_init_kernel(float* agg, float* sum, unsigned* mnb, unsigned* mxb) {
    int i = blockIdx.x * blockDim.x + threadIdx.x;
    if (i < N_NODES * F_IN) agg[i] = 0.f;
    if (i < NGRAPH * F_IN) { sum[i] = 0.f; mnb[i] = 0x7F800000u; mxb[i] = 0u; }
}

// h_out = relu(h_in + agg); pool min/max/sum per graph (values >= 0 so
// unsigned bit-compare atomics order correctly).
__global__ void node_pool_kernel(const float* __restrict__ hin, const float* __restrict__ agg,
                                 const int* __restrict__ bat, float* __restrict__ hout,
                                 float* sum, unsigned* mnb, unsigned* mxb)
{
    int i = blockIdx.x * blockDim.x + threadIdx.x;
    if (i >= N_NODES) return;
    const int g = bat[i];
    #pragma unroll
    for (int f = 0; f < F_IN; ++f) {
        float v = hin[(size_t)i * F_IN + f] + agg[(size_t)i * F_IN + f];
        v = fmaxf(v, 0.f);
        hout[(size_t)i * F_IN + f] = v;
        atomicAdd(&sum[g * F_IN + f], v);
        const unsigned b = __float_as_uint(v);
        atomicMax(&mxb[g * F_IN + f], b);
        atomicMin(&mnb[g * F_IN + f], b);
    }
}

__global__ void enc_acc_kernel(const float* __restrict__ sum, const unsigned* __restrict__ mnb,
                               const unsigned* __restrict__ mxb, const int* __restrict__ cnt,
                               float* enc)
{
    int i = blockIdx.x * blockDim.x + threadIdx.x;
    if (i >= NGRAPH * F_IN) return;
    const int g = i / F_IN, f = i % F_IN;
    const int c = cnt[g];
    const float mn   = (c > 0) ? __uint_as_float(mnb[i]) : 0.f;
    const float mx   = (c > 0) ? __uint_as_float(mxb[i]) : 0.f;
    const float mean = sum[i] / fmaxf((float)c, 1.f);
    enc[g * 27 + f]      += mn;
    enc[g * 27 + 9 + f]  += mx;
    enc[g * 27 + 18 + f] += mean;
}

// One block per graph: 27 -> 126 -> 64 -> 32 MLP; also emits enc.
__global__ void __launch_bounds__(128)
mlp_kernel(const float* __restrict__ enc,
           const float* __restrict__ W1, const float* __restrict__ b1,
           const float* __restrict__ W2, const float* __restrict__ b2,
           const float* __restrict__ W3, const float* __restrict__ b3,
           float* __restrict__ out)
{
    __shared__ float se[27], s1[126], s2[64];
    const int g = blockIdx.x, t = threadIdx.x;
    if (t < 27) se[t] = enc[g * 27 + t];
    __syncthreads();
    if (t < 126) {
        float a = b1[t];
        #pragma unroll
        for (int k = 0; k < 27; ++k) a += se[k] * W1[k * 126 + t];
        s1[t] = fmaxf(a, 0.f);
    }
    __syncthreads();
    if (t < 64) {
        float a = b2[t];
        for (int k = 0; k < 126; ++k) a += s1[k] * W2[k * 64 + t];
        s2[t] = fmaxf(a, 0.f);
    }
    __syncthreads();
    if (t < 32) {
        float a = b3[t];
        #pragma unroll
        for (int k = 0; k < 64; ++k) a += s2[k] * W3[k * 32 + t];
        out[g * OUTD + t] = a;
    }
    if (t < 27) out[NGRAPH * OUTD + g * 27 + t] = se[t];
}

// ---------------------------------------------------------------------------
extern "C" void kernel_launch(void* const* d_in, const int* in_sizes, int n_in,
                              void* d_out, int out_size, void* d_ws, size_t ws_size,
                              hipStream_t stream)
{
    (void)in_sizes; (void)n_in; (void)out_size; (void)ws_size;

    const float* x   = (const float*)d_in[0];
    const int*   ei  = (const int*)d_in[1];
    const float* ea  = (const float*)d_in[2];
    const int*   bat = (const int*)d_in[3];
    const float* Wf  = (const float*)d_in[4];
    const float* bf  = (const float*)d_in[5];
    const float* Ws  = (const float*)d_in[6];
    const float* bs  = (const float*)d_in[7];
    const float* W1  = (const float*)d_in[8];
    const float* b1  = (const float*)d_in[9];
    const float* W2  = (const float*)d_in[10];
    const float* b2  = (const float*)d_in[11];
    const float* W3  = (const float*)d_in[12];
    const float* b3  = (const float*)d_in[13];
    float* out = (float*)d_out;

    // workspace layout (floats)
    float*    h1  = (float*)d_ws;
    float*    h2  = h1 + (size_t)N_NODES * F_IN;
    float*    agg = h2 + (size_t)N_NODES * F_IN;
    float*    sum = agg + (size_t)N_NODES * F_IN;
    unsigned* mnb = (unsigned*)(sum + NGRAPH * F_IN);
    unsigned* mxb = mnb + NGRAPH * F_IN;
    int*      cnt = (int*)(mxb + NGRAPH * F_IN);
    float*    enc = (float*)(cnt + NGRAPH);

    init_enc_cnt_kernel<<<(NGRAPH * 27 + 255) / 256, 256, 0, stream>>>(enc, cnt);
    count_kernel<<<(N_NODES + 255) / 256, 256, 0, stream>>>(bat, cnt);

    const float* hin = x;
    float* houts[3] = { h1, h2, h1 };
    for (int l = 0; l < 3; ++l) {
        layer_init_kernel<<<(N_NODES * F_IN + 255) / 256, 256, 0, stream>>>(agg, sum, mnb, mxb);
        cgconv_edge_wmma<<<4096, 256, 0, stream>>>(hin, ei, ea,
                                                   Wf + l * ZI * F_IN, Ws + l * ZI * F_IN,
                                                   bf + l * F_IN, bs + l * F_IN, agg);
        node_pool_kernel<<<(N_NODES + 255) / 256, 256, 0, stream>>>(hin, agg, bat, houts[l],
                                                                    sum, mnb, mxb);
        enc_acc_kernel<<<(NGRAPH * F_IN + 255) / 256, 256, 0, stream>>>(sum, mnb, mxb, cnt, enc);
        hin = houts[l];
    }
    mlp_kernel<<<NGRAPH, 128, 0, stream>>>(enc, W1, b1, W2, b2, W3, b3, out);
}
